// PerTokenFFN_34076270526842
// MI455X (gfx1250) — matrix operand order repaired and checked by reading, not compiled, not run on hardware
//
#include <hip/hip_runtime.h>

// Per-token-slot expert FFN for MI455X (gfx1250, wave32, WMMA).
// out[b,t,:] = gelu(x[b,t,:] @ W1[t] + b1[t]) @ W2[t] + b2[t]
// B=1024, T=64, D=512, H=2048. Fully fused via H-chunking (h never hits HBM).

typedef _Float16 v16h __attribute__((ext_vector_type(16)));
typedef _Float16 v8h  __attribute__((ext_vector_type(8)));
typedef _Float16 v4h  __attribute__((ext_vector_type(4)));
typedef float    v8f  __attribute__((ext_vector_type(8)));

constexpr int Bsz = 1024;
constexpr int T   = 64;
constexpr int D   = 512;
constexpr int H   = 2048;
constexpr int MT  = 64;          // rows (b) per workgroup
constexpr int HC  = 32;          // H-chunk size
constexpr int NCHUNK = H / HC;   // 64

// LDS byte offsets (all data stored in WMMA fragment order, f16)
constexpr int XA_OFF  = 0;                     // [mt=4][kb=16][lane=32][j=16] = 65536 B
constexpr int WB1_OFF = 65536;                 // [kb=16][nt=2][lane=32][j=16] = 32768 B
constexpr int WB2_OFF = 65536 + 32768;         // [nt=32][lane=32][j=16]       = 32768 B
constexpr int HA_OFF  = 65536 + 32768 + 32768; // [mt=4][lane=32][j=16]        =  4096 B
constexpr int LDS_BYTES = HA_OFF + 4096;       // 135168 (<= 320KB; 2 WGs/WGP)

__device__ __forceinline__ float fast_tanh(float u) {
    // overflow-safe: e^{2|u|} -> inf  =>  result -> 1
    float eu = __expf(2.0f * fabsf(u));            // v_exp_f32 (TRANS, co-executes)
    float th = 1.0f - __fdividef(2.0f, eu + 1.0f);
    return copysignf(th, u);
}

extern "C" __global__ __launch_bounds__(256)
void ffn_moe_wmma(const float* __restrict__ X,  const float* __restrict__ W1,
                  const float* __restrict__ B1, const float* __restrict__ W2,
                  const float* __restrict__ B2, float* __restrict__ OUT)
{
    extern __shared__ char smem[];
    _Float16* xA  = (_Float16*)(smem + XA_OFF);
    _Float16* wB1 = (_Float16*)(smem + WB1_OFF);
    _Float16* wB2 = (_Float16*)(smem + WB2_OFF);
    _Float16* hA  = (_Float16*)(smem + HA_OFF);

    const int tid  = threadIdx.x;
    const int lane = tid & 31;
    const int wv   = tid >> 5;        // 0..7
    const int r    = wv & 3;          // wave M-tile (16 rows each)
    const int c    = wv >> 2;         // wave N-half (0..1)
    const int t    = blockIdx.y;      // expert / token slot
    const int b0   = blockIdx.x * MT; // first batch row of this tile

    // Uniform scalar bases; all per-lane offsets are 32-bit (slices < 4GB)
    const float* __restrict__ Xb  = X  + ((size_t)b0 * T + t) * D;  // + ml*T*D + d
    const float* __restrict__ W1t = W1 + (size_t)t * D * H;         // + d*H + h
    const float* __restrict__ B1t = B1 + (size_t)t * H;
    const float* __restrict__ W2t = W2 + (size_t)t * H * D;         // + h*D + d
    const float* __restrict__ B2t = B2 + (size_t)t * D;
    float*       __restrict__ Ob  = OUT + ((size_t)b0 * T + t) * D; // + m*T*D + d

    // ---------- load x tile (64x512 fp32) -> f16 A-fragments ----------
    // A-frag map (16x32 f16): lane = half*16 + (m&15), half=(k>>3)&1, j=(k&7)+8*(k>>4)
    for (int i = 0; i < (MT * D / 4) / 256; ++i) {   // 32 float4 per thread
        int q  = tid + 256 * i;                      // 0..8191
        int ml = q >> 7;                             // row in tile 0..63
        int d0 = (q & 127) << 2;                     // col 0..508 step 4
        const float4 v = *(const float4*)(Xb + ml * (T * D) + d0);
        int kb = d0 >> 5, ko = d0 & 31;
        int hf = (ko >> 3) & 1;
        int j  = (ko & 7) + ((ko >> 4) << 3);        // d0%4==0 => 4 contiguous j
        int ln = hf * 16 + (ml & 15);
        int mt = ml >> 4;
        v4h h4 = { (_Float16)v.x, (_Float16)v.y, (_Float16)v.z, (_Float16)v.w };
        *(v4h*)(xA + (((mt * 16 + kb) * 32 + ln) * 16 + j)) = h4;
    }

    v8f oacc[16];
    #pragma unroll
    for (int n = 0; n < 16; ++n)
        oacc[n] = v8f{0.f,0.f,0.f,0.f,0.f,0.f,0.f,0.f};

    const v16h* xaF  = (const v16h*)xA;
    const v16h* wb1F = (const v16h*)wB1;
    const v16h* wb2F = (const v16h*)wB2;
    const v16h* haF  = (const v16h*)hA;

    for (int hc = 0; hc < NCHUNK; ++hc) {
        const int hb = hc * HC;

        // ---- W1 chunk (512 x 32 fp32) -> f16 B-fragments ----
        // B-frag map (32x16 f16): lane = ((k>>3)&1)*16 + (n&15), j = (k&7)+8*(k>>4).
        // j is contiguous in K: a task of 8 consecutive K-rows x 4 cols packs
        // each column into one 16B-aligned v8h -> single ds_store_b128.
        #pragma unroll
        for (int i = 0; i < 2; ++i) {
            int s  = tid + 256 * i;                  // task 0..511
            int k0 = (s >> 3) << 3;                  // K row group 0..504 step 8
            int n0 = (s & 7) << 2;                   // N col 0..28 step 4
            float rv[8][4];
            #pragma unroll
            for (int e = 0; e < 8; ++e) {
                const float4 v = *(const float4*)(W1t + (k0 + e) * H + hb + n0);
                rv[e][0] = v.x; rv[e][1] = v.y; rv[e][2] = v.z; rv[e][3] = v.w;
            }
            int kb = k0 >> 5;
            int hf = (k0 >> 3) & 1;
            int j0 = ((k0 >> 4) & 1) << 3;           // 0 or 8
            #pragma unroll
            for (int u = 0; u < 4; ++u) {
                int n = n0 + u;
                v8h p = { (_Float16)rv[0][u], (_Float16)rv[1][u],
                          (_Float16)rv[2][u], (_Float16)rv[3][u],
                          (_Float16)rv[4][u], (_Float16)rv[5][u],
                          (_Float16)rv[6][u], (_Float16)rv[7][u] };
                *(v8h*)(wB1 + (((kb * 2 + (n >> 4)) * 32 + hf * 16 + (n & 15)) * 16 + j0)) = p;
            }
        }
        // ---- W2 chunk (32 x 512 fp32) -> f16 B-fragments (same trick) ----
        #pragma unroll
        for (int i = 0; i < 2; ++i) {
            int s  = tid + 256 * i;                  // task 0..511
            int k0 = (s >> 7) << 3;                  // K row group 0,8,16,24
            int n0 = (s & 127) << 2;                 // N col 0..508 step 4
            float rv[8][4];
            #pragma unroll
            for (int e = 0; e < 8; ++e) {
                const float4 v = *(const float4*)(W2t + (hb + k0 + e) * D + n0);
                rv[e][0] = v.x; rv[e][1] = v.y; rv[e][2] = v.z; rv[e][3] = v.w;
            }
            int hf = (k0 >> 3) & 1;
            int j0 = ((k0 >> 4) & 1) << 3;           // 0 or 8
            #pragma unroll
            for (int u = 0; u < 4; ++u) {
                int n = n0 + u;
                v8h p = { (_Float16)rv[0][u], (_Float16)rv[1][u],
                          (_Float16)rv[2][u], (_Float16)rv[3][u],
                          (_Float16)rv[4][u], (_Float16)rv[5][u],
                          (_Float16)rv[6][u], (_Float16)rv[7][u] };
                *(v8h*)(wB2 + (((n >> 4) * 32 + hf * 16 + (n & 15)) * 16 + j0)) = p;
            }
        }
        __syncthreads();

        // ---- GEMM1: h(64x32) = x_tile(64x512) @ W1c(512x32), fp32 accum ----
        // software-pipelined: prefetch frag kb+1 before issuing WMMA kb
        v8f hacc = v8f{0.f,0.f,0.f,0.f,0.f,0.f,0.f,0.f};
        v16h a_cur = xaF[(r * 16 + 0) * 32 + lane];
        v16h b_cur = wb1F[(0 * 2 + c) * 32 + lane];
        #pragma unroll
        for (int kb = 0; kb < 16; ++kb) {
            v16h a_nxt = a_cur, b_nxt = b_cur;
            if (kb < 15) {
                a_nxt = xaF[(r * 16 + kb + 1) * 32 + lane];
                b_nxt = wb1F[((kb + 1) * 2 + c) * 32 + lane];
            }
            hacc = __builtin_amdgcn_wmma_f32_16x16x32_f16(
                false, a_cur, false, b_cur, (short)0, hacc, false, false);
            a_cur = a_nxt; b_cur = b_nxt;
        }

        // ---- bias + tanh-GELU in registers, write f16 A-frags of h ----
        {
            int nloc   = 16 * c + (lane & 15);           // h-col within chunk
            float bias = B1t[hb + nloc];
            int mh = lane >> 4;
            int hf = (nloc >> 3) & 1;
            int j  = (nloc & 7) + ((nloc >> 4) << 3);
            #pragma unroll
            for (int g = 0; g < 8; ++g) {
                float xv = hacc[g] + bias;
                float u  = 0.7978845608028654f * xv * (1.0f + 0.044715f * xv * xv);
                float gv = 0.5f * xv * (1.0f + fast_tanh(u));
                hA[((r * 32 + hf * 16 + (g + 8 * mh)) * 16 + j)] = (_Float16)gv;
            }
        }
        __syncthreads();

        // ---- GEMM2: out_tile += gelu_h(64x32) @ W2c(32x512), pipelined ----
        v16h a2 = haF[r * 32 + lane];
        v16h w_cur = wb2F[(c * 16 + 0) * 32 + lane];
        #pragma unroll
        for (int nt = 0; nt < 16; ++nt) {
            v16h w_nxt = w_cur;
            if (nt < 15) w_nxt = wb2F[(c * 16 + nt + 1) * 32 + lane];
            oacc[nt] = __builtin_amdgcn_wmma_f32_16x16x32_f16(
                false, a2, false, w_cur, (short)0, oacc[nt], false, false);
            w_cur = w_nxt;
        }
        __syncthreads();   // before next chunk overwrites wB1/wB2
    }

    // ---- epilogue: + b2, store fp32 ----
    #pragma unroll
    for (int nt = 0; nt < 16; ++nt) {
        int d = 256 * c + 16 * nt + (lane & 15);
        float bias2 = B2t[d];
        #pragma unroll
        for (int g = 0; g < 8; ++g) {
            int m = 16 * r + g + 8 * (lane >> 4);
            Ob[m * (T * D) + d] = oacc[nt][g] + bias2;
        }
    }
}

extern "C" void kernel_launch(void* const* d_in, const int* in_sizes, int n_in,
                              void* d_out, int out_size, void* d_ws, size_t ws_size,
                              hipStream_t stream) {
    (void)in_sizes; (void)n_in; (void)out_size; (void)d_ws; (void)ws_size;
    const float* X  = (const float*)d_in[0];
    const float* W1 = (const float*)d_in[1];
    const float* B1 = (const float*)d_in[2];
    const float* W2 = (const float*)d_in[3];
    const float* B2 = (const float*)d_in[4];
    float* OUT = (float*)d_out;

    // allow >64KB dynamic LDS (gfx1250 WGP has 320KB)
    hipFuncSetAttribute((const void*)ffn_moe_wmma,
                        hipFuncAttributeMaxDynamicSharedMemorySize, LDS_BYTES);

    dim3 grid(Bsz / MT, T, 1);   // (16, 64): blocks sharing an expert run adjacently -> L2 reuse
    dim3 block(256, 1, 1);       // 8 waves (wave32), 4x2 wave tiling over 64x512 output
    ffn_moe_wmma<<<grid, block, LDS_BYTES, stream>>>(X, W1, B1, W2, B2, OUT);
}
